// LocallyGroupedAttnRPEContext_70978629534048
// MI455X (gfx1250) — matrix operand order repaired and checked by reading, not compiled, not run on hardware
//
#include <hip/hip_runtime.h>
#include <hip/hip_bf16.h>

// ---------------------------------------------------------------------------
// LocallyGroupedAttnRPEContext on MI455X (gfx1250), wave32 + WMMA f16.
// Pipeline: weights->f16^T | build(ctx-proj WMMA + PE + window partition) |
//           q/k/v proj WMMA | per-(window,head) attention WMMA | out-proj WMMA
// B operands pre-transposed [N][K] in global (L2-resident, reused 1536x) so
// every fragment load is 2 x b128; A operands staged/made in LDS.
// ---------------------------------------------------------------------------

typedef __attribute__((ext_vector_type(16))) _Float16 v16h;
typedef __attribute__((ext_vector_type(8)))  _Float16 v8h;
typedef __attribute__((ext_vector_type(8)))  float    v8f;

#define WMMA16(a, b, c) \
  __builtin_amdgcn_wmma_f32_16x16x32_f16(false, (a), false, (b), (short)0, (c), false, false)

#define DIMC 256
#define NHEAD 8
#define WSZ 8
#define VCH 64
#define CQK 320
#define HDIM 32
#define HH 94
#define WW_ 126
#define BB 8
#define NWH 12
#define NWW 16
#define NWIN (BB * NWH * NWW)   /* 1536 windows  */
#define TOK 64                  /* tokens/window */
#define NT ((size_t)NWIN * TOK) /* 98304         */

// ---------------- WMMA fragment loaders (ISA 7.12.2 layouts) ----------------
// A fragment 16x32 f16 from row-major [M][K]: two contiguous 8-half runs.
__device__ __forceinline__ v16h frag_a_rm(const _Float16* s, int ld, int row0, int k0) {
  int lane = threadIdx.x & 31;
  int m = row0 + (lane & 15);
  int kb = k0 + 8 * (lane >> 4);
  const _Float16* base = s + m * ld + kb;
  v8h lo = *(const v8h*)(base);
  v8h hi = *(const v8h*)(base + 16);
  v16h a;
#pragma unroll
  for (int i = 0; i < 8; ++i) { a[i] = lo[i]; a[i + 8] = hi[i]; }
  return a;
}

// B fragment 32x16 f16 from row-major [N][K] (transposed source): two
// contiguous 8-half runs per lane -> two b128 loads.
__device__ __forceinline__ v16h frag_b_nk(const _Float16* s, int ld, int k0, int col0) {
  int lane = threadIdx.x & 31;
  int n = col0 + (lane & 15);
  int kb = k0 + 8 * (lane >> 4);
  const _Float16* base = s + n * ld + kb;
  v8h lo = *(const v8h*)(base);
  v8h hi = *(const v8h*)(base + 16);
  v16h b;
#pragma unroll
  for (int i = 0; i < 8; ++i) { b[i] = lo[i]; b[i + 8] = hi[i]; }
  return b;
}

// Linear sine positional embedding (FlowFormer, dim=CQK, f[i]=i, 1/200 norm).
__device__ __forceinline__ float pe_val(int row, int col, int ch) {
  const float kk = 3.14f / 200.0f;
  if (ch < 80)  return __sinf(kk * (float)col * (float)ch);
  if (ch < 160) return __cosf(kk * (float)col * (float)(ch - 80));
  if (ch < 240) return __sinf(kk * (float)row * (float)(ch - 160));
  return __cosf(kk * (float)row * (float)(ch - 240));
}

// ------------------------------- kernels ------------------------------------

// Transpose-convert W [K][N] f32 -> W^T [N][K] f16.
__global__ void cvt_T(const float* __restrict__ w, _Float16* __restrict__ wT,
                      int K, int N) {
  int i = blockIdx.x * blockDim.x + threadIdx.x;  // i = k*N + n
  if (i < K * N) {
    int k = i / N, n = i - k * N;
    wT[n * K + k] = (_Float16)w[i];
  }
}

// One block per window (256 threads = 8 waves).
// ctx = gather(context)[64x256] @ Wc[256x64] (WMMA, K streamed in 32-slices,
// B fragments straight from global WcT), then writes xqk = pad(x||ctx+bc)+PE
// and xg = pad(x) as f16.
__global__ void __launch_bounds__(256)
build_inputs_kernel(const float* __restrict__ x, const float* __restrict__ ctx,
                    const _Float16* __restrict__ wcT, const float* __restrict__ bc,
                    _Float16* __restrict__ xqk, _Float16* __restrict__ xg) {
  __shared__ alignas(16) _Float16 sA[64 * 32];  // ctx K-slice ( 4 KB)
  __shared__ alignas(16) float    sC[64 * 64];  // ctx result  (16 KB)

  const int win = blockIdx.x;
  const int b = win / (NWH * NWW);
  const int wh = (win / NWW) % NWH;
  const int ww = win % NWW;
  const int tid = threadIdx.x, lane = tid & 31, wave = tid >> 5;

  const int rt = wave >> 1;           // row tile 0..3
  const int ct0 = (wave & 1) * 2;     // col tile pair {0,1} or {2,3}
  v8f acc0 = {}, acc1 = {};
  for (int k0 = 0; k0 < 256; k0 += 32) {
    __syncthreads();
    // gather context slice, branch-free (clamped address + select)
    for (int e = tid; e < 64 * 32; e += 256) {
      int t = e & 63, kc = e >> 6;
      int gy = wh * WSZ + (t >> 3), gx = ww * WSZ + (t & 7);
      bool valid = (gy < HH) && (gx < WW_);
      size_t idx = (((size_t)b * 256 + (size_t)(k0 + kc)) * HH + (valid ? gy : 0)) * WW_ +
                   (valid ? gx : 0);
      float vv = ctx[idx];
      sA[t * 32 + kc] = (_Float16)(valid ? vv : 0.0f);
    }
    __syncthreads();
    v16h a  = frag_a_rm(sA, 32, rt * 16, 0);
    v16h b0 = frag_b_nk(wcT, 256, k0, ct0 * 16);
    v16h b1 = frag_b_nk(wcT, 256, k0, ct0 * 16 + 16);
    acc0 = WMMA16(a, b0, acc0);
    acc1 = WMMA16(a, b1, acc1);
  }
  {
    int n = lane & 15, g = lane >> 4;
#pragma unroll
    for (int j = 0; j < 8; ++j) {
      int m = rt * 16 + j + 8 * g;
      sC[m * 64 + ct0 * 16 + n]      = acc0[j];
      sC[m * 64 + ct0 * 16 + 16 + n] = acc1[j];
    }
  }
  __syncthreads();

  const size_t base = (size_t)win * TOK;
  for (int e = tid; e < TOK * CQK; e += 256) {
    int t = e / CQK, ch = e - t * CQK;
    int r = t >> 3, c = t & 7;
    int gy = wh * WSZ + r, gx = ww * WSZ + c;
    bool valid = (gy < HH) && (gx < WW_);
    size_t tokidx = (size_t)b * (HH * WW_) + (size_t)(valid ? gy : 0) * WW_ + (valid ? gx : 0);
    float v;
    if (ch < DIMC) {
      float xv = x[tokidx * DIMC + ch];
      v = valid ? xv : 0.0f;
    } else {
      v = valid ? (sC[t * 64 + (ch - DIMC)] + bc[ch - DIMC]) : 0.0f;
    }
    v += pe_val(r, c, ch);
    xqk[(base + t) * CQK + ch] = (_Float16)v;
  }
  for (int e = tid; e < TOK * DIMC; e += 256) {
    int t = e >> 8, ch = e & 255;
    int gy = wh * WSZ + (t >> 3), gx = ww * WSZ + (t & 7);
    bool valid = (gy < HH) && (gx < WW_);
    size_t tokidx = (size_t)b * (HH * WW_) + (size_t)(valid ? gy : 0) * WW_ + (valid ? gx : 0);
    float xv = x[tokidx * DIMC + ch];
    xg[(base + t) * DIMC + ch] = (_Float16)(valid ? xv : 0.0f);
  }
}

// grid (NWIN, 3): y==0 -> q (scaled), y==1 -> k, y==2 -> v. 256 threads.
// B fragments straight from global transposed weights (L2-resident).
__global__ void __launch_bounds__(256)
proj_kernel(const _Float16* __restrict__ xqk, const _Float16* __restrict__ xg,
            const _Float16* __restrict__ wqT, const _Float16* __restrict__ wkT,
            const _Float16* __restrict__ wvT,
            const float* __restrict__ bq, const float* __restrict__ bk,
            const float* __restrict__ bv,
            _Float16* __restrict__ q, _Float16* __restrict__ k, _Float16* __restrict__ v) {
  __shared__ alignas(16) _Float16 sA[64 * CQK];  // 40 KB (max Kdim)

  const int win = blockIdx.x;
  const int which = blockIdx.y;
  const _Float16* in  = (which == 2) ? xg : xqk;
  const int kdim      = (which == 2) ? DIMC : CQK;
  const _Float16* wT  = (which == 0) ? wqT : (which == 1 ? wkT : wvT);
  const float* bias   = (which == 0) ? bq : (which == 1 ? bk : bv);
  _Float16* out       = (which == 0) ? q : (which == 1 ? k : v);
  const float scale   = (which == 0) ? rsqrtf((float)HDIM) : 1.0f;

  const int tid = threadIdx.x, lane = tid & 31, wave = tid >> 5;
  const size_t base = (size_t)win * TOK;

  // vector copy of activations into LDS (contiguous both sides)
  for (int e = tid; e < (TOK * kdim) / 8; e += 256)
    ((v8h*)sA)[e] = ((const v8h*)(in + base * kdim))[e];
  __syncthreads();

  v8f acc[4][2] = {};
  const int ct0 = wave * 2;  // each wave owns 2 of 16 col tiles, all 4 row tiles
  for (int k0 = 0; k0 < kdim; k0 += 32) {
    v16h b0 = frag_b_nk(wT, kdim, k0, ct0 * 16);
    v16h b1 = frag_b_nk(wT, kdim, k0, ct0 * 16 + 16);
#pragma unroll
    for (int r = 0; r < 4; ++r) {
      v16h a = frag_a_rm(sA, kdim, r * 16, k0);
      acc[r][0] = WMMA16(a, b0, acc[r][0]);
      acc[r][1] = WMMA16(a, b1, acc[r][1]);
    }
  }
  const int n = lane & 15, g = lane >> 4;
#pragma unroll
  for (int r = 0; r < 4; ++r)
#pragma unroll
    for (int cc = 0; cc < 2; ++cc) {
      int col = (ct0 + cc) * 16 + n;
      float bs = bias[col];
#pragma unroll
      for (int j = 0; j < 8; ++j) {
        int m = r * 16 + j + 8 * g;
        out[(base + m) * 256 + col] = (_Float16)((acc[r][cc][j] + bs) * scale);
      }
    }
}

// One block per (window, head); 64 threads (2 waves).
// q/k fragments straight from global; V transposed via LDS for P@V.
__global__ void __launch_bounds__(64)
attn_kernel(const _Float16* __restrict__ q, const _Float16* __restrict__ k,
            const _Float16* __restrict__ v, _Float16* __restrict__ ow) {
  __shared__ alignas(16) _Float16 svT[32 * 64];  // [d][tok]   4 KB
  __shared__ alignas(16) float    sS[64 * 64];   // scores    16 KB
  __shared__ alignas(16) _Float16 sP[64 * 64];   // probs      8 KB

  const int win = blockIdx.x >> 3;
  const int h = blockIdx.x & 7;
  const int tid = threadIdx.x, lane = tid & 31, wave = tid >> 5;
  const size_t base = (size_t)win * TOK;
  const _Float16* qh = q + base * 256 + (size_t)h * 32;  // [tok][256] slice
  const _Float16* kh = k + base * 256 + (size_t)h * 32;

  // stage V transposed: svT[d][tok]
  for (int e8 = tid * 8; e8 < 64 * 32; e8 += 64 * 8) {
    int t = e8 >> 5, d0 = e8 & 31;
    v8h v8 = *(const v8h*)(v + (base + t) * 256 + (size_t)h * 32 + d0);
#pragma unroll
    for (int i = 0; i < 8; ++i) svT[(d0 + i) * 64 + t] = v8[i];
  }
  __syncthreads();

  const int n = lane & 15, g = lane >> 4;
  // scores = q @ k^T (scale folded into q); A rows and B rows stride 256
#pragma unroll
  for (int rr = 0; rr < 2; ++rr) {
    int rt = wave * 2 + rr;
    v16h a = frag_a_rm(qh, 256, rt * 16, 0);
#pragma unroll
    for (int ct = 0; ct < 4; ++ct) {
      v16h bf = frag_b_nk(kh, 256, 0, ct * 16);
      v8f c = {};
      c = WMMA16(a, bf, c);
#pragma unroll
      for (int j = 0; j < 8; ++j)
        sS[(rt * 16 + j + 8 * g) * 64 + ct * 16 + n] = c[j];
    }
  }
  __syncthreads();

  // softmax: one row per thread
  {
    int r = tid;
    float mx = -1e30f;
    for (int c0 = 0; c0 < 64; ++c0) mx = fmaxf(mx, sS[r * 64 + c0]);
    float sum = 0.0f;
    for (int c0 = 0; c0 < 64; ++c0) {
      float e = __expf(sS[r * 64 + c0] - mx);
      sS[r * 64 + c0] = e;
      sum += e;
    }
    float inv = 1.0f / sum;
    for (int c0 = 0; c0 < 64; ++c0) sP[r * 64 + c0] = (_Float16)(sS[r * 64 + c0] * inv);
  }
  __syncthreads();

  // out = P @ V   (B source = svT [d][tok], i.e. [N][K])
#pragma unroll
  for (int rr = 0; rr < 2; ++rr) {
    int rt = wave * 2 + rr;
#pragma unroll
    for (int ct = 0; ct < 2; ++ct) {
      v8f acc = {};
#pragma unroll
      for (int k0 = 0; k0 < 64; k0 += 32) {
        v16h a = frag_a_rm(sP, 64, rt * 16, k0);
        v16h bf = frag_b_nk(svT, 64, k0, ct * 16);
        acc = WMMA16(a, bf, acc);
      }
#pragma unroll
      for (int j = 0; j < 8; ++j) {
        int m = rt * 16 + j + 8 * g;
        ow[(base + m) * 256 + (size_t)h * 32 + ct * 16 + n] = (_Float16)acc[j];
      }
    }
  }
}

// One block per window; out = ow @ Wo + bo scattered into [B,N,256] fp32.
__global__ void __launch_bounds__(256)
outproj_kernel(const _Float16* __restrict__ ow, const _Float16* __restrict__ woT,
               const float* __restrict__ bo, float* __restrict__ out) {
  __shared__ alignas(16) _Float16 sA[64 * 256];  // 32 KB

  const int win = blockIdx.x;
  const int b = win / (NWH * NWW);
  const int wh = (win / NWW) % NWH;
  const int ww = win % NWW;
  const int tid = threadIdx.x, lane = tid & 31, wave = tid >> 5;
  const size_t base = (size_t)win * TOK;

  for (int e = tid; e < (64 * 256) / 8; e += 256)
    ((v8h*)sA)[e] = ((const v8h*)(ow + base * 256))[e];
  __syncthreads();

  v8f acc[4][2] = {};
  const int ct0 = wave * 2;
  for (int k0 = 0; k0 < 256; k0 += 32) {
    v16h b0 = frag_b_nk(woT, 256, k0, ct0 * 16);
    v16h b1 = frag_b_nk(woT, 256, k0, ct0 * 16 + 16);
#pragma unroll
    for (int r = 0; r < 4; ++r) {
      v16h a = frag_a_rm(sA, 256, r * 16, k0);
      acc[r][0] = WMMA16(a, b0, acc[r][0]);
      acc[r][1] = WMMA16(a, b1, acc[r][1]);
    }
  }
  const int n = lane & 15, g = lane >> 4;
#pragma unroll
  for (int r = 0; r < 4; ++r)
#pragma unroll
    for (int cc = 0; cc < 2; ++cc) {
      int col = (ct0 + cc) * 16 + n;
      float bs = bo[col];
#pragma unroll
      for (int j = 0; j < 8; ++j) {
        int t = r * 16 + j + 8 * g;
        int gy = wh * WSZ + (t >> 3), gx = ww * WSZ + (t & 7);
        if (gy < HH && gx < WW_)
          out[((size_t)b * (HH * WW_) + (size_t)gy * WW_ + gx) * 256 + col] =
              acc[r][cc][j] + bs;
      }
    }
}

// ------------------------------- launcher -----------------------------------

extern "C" void kernel_launch(void* const* d_in, const int* in_sizes, int n_in,
                              void* d_out, int out_size, void* d_ws, size_t ws_size,
                              hipStream_t stream) {
  const float* x    = (const float*)d_in[0];
  const float* ctx  = (const float*)d_in[1];
  const float* Wc   = (const float*)d_in[2];
  const float* bc   = (const float*)d_in[3];
  const float* Wq   = (const float*)d_in[4];
  const float* bq   = (const float*)d_in[5];
  const float* Wk   = (const float*)d_in[6];
  const float* bk   = (const float*)d_in[7];
  const float* Wv   = (const float*)d_in[8];
  const float* bv   = (const float*)d_in[9];
  const float* Wo   = (const float*)d_in[10];
  const float* bo   = (const float*)d_in[11];
  float* out = (float*)d_out;

  // f16 workspace layout (element offsets); all transposed weights are [N][K]
  _Float16* ws = (_Float16*)d_ws;
  const size_t O_WCT = 0;                               // [64][256]
  const size_t O_WQT = O_WCT + (size_t)64 * 256;        // [256][320]
  const size_t O_WKT = O_WQT + (size_t)256 * CQK;       // [256][320]
  const size_t O_WVT = O_WKT + (size_t)256 * CQK;       // [256][256]
  const size_t O_WOT = O_WVT + (size_t)256 * 256;       // [256][256]
  const size_t O_XQK = O_WOT + (size_t)256 * 256;
  const size_t O_XG  = O_XQK + NT * CQK;
  const size_t O_Q   = O_XG + NT * 256;
  const size_t O_K   = O_Q + NT * 256;
  const size_t O_V   = O_K + NT * 256;
  const size_t O_OW  = O_V + NT * 256;

  _Float16* wWcT = ws + O_WCT;
  _Float16* wWqT = ws + O_WQT;
  _Float16* wWkT = ws + O_WKT;
  _Float16* wWvT = ws + O_WVT;
  _Float16* wWoT = ws + O_WOT;
  _Float16* xqk  = ws + O_XQK;
  _Float16* xg   = ws + O_XG;
  _Float16* qb   = ws + O_Q;
  _Float16* kb_  = ws + O_K;
  _Float16* vb   = ws + O_V;
  _Float16* owb  = ws + O_OW;

  auto cvtT = [&](const float* src, _Float16* dstT, int K, int N) {
    cvt_T<<<(K * N + 255) / 256, 256, 0, stream>>>(src, dstT, K, N);
  };
  cvtT(Wc, wWcT, 256, 64);
  cvtT(Wq, wWqT, CQK, 256);
  cvtT(Wk, wWkT, CQK, 256);
  cvtT(Wv, wWvT, 256, 256);
  cvtT(Wo, wWoT, 256, 256);

  build_inputs_kernel<<<NWIN, 256, 0, stream>>>(x, ctx, wWcT, bc, xqk, xg);

  proj_kernel<<<dim3(NWIN, 3), 256, 0, stream>>>(xqk, xg, wWqT, wWkT, wWvT,
                                                 bq, bk, bv, qb, kb_, vb);

  attn_kernel<<<NWIN * NHEAD, 64, 0, stream>>>(qb, kb_, vb, owb);

  outproj_kernel<<<NWIN, 256, 0, stream>>>(owb, wWoT, bo, out);
}
